// GraphNeuralNetwork_1503238553636
// MI455X (gfx1250) — matrix operand order repaired and checked by reading, not compiled, not run on hardware
//
#include <hip/hip_runtime.h>

typedef __attribute__((ext_vector_type(16))) _Float16 v16h;
typedef __attribute__((ext_vector_type(8)))  float    v8f;
typedef int v4i_ __attribute__((vector_size(16)));

#define NW    5
#define B_    16
#define S_    25
#define Q_    75
#define N_    26
#define NN    676          /* N*N                      */
#define RPQ   10816        /* B*N*N rows per q-slice   */
#define TPQ   676          /* 16-row tiles per q       */
#define QC    15           /* queries per chunk        */
#define NCHUNK 5
#define FSTR  256          /* node feature stride      */
#define LREL  0.01f
#define EPS_  1e-5f
#define SSOFF (Q_*192)     /* sumsq / shift plane offset */

// ---- gfx1250 async global->LDS path (guarded: falls back to sync staging) --
#if defined(__has_builtin)
#if __has_builtin(__builtin_amdgcn_global_load_async_to_lds_b128) && \
    __has_builtin(__builtin_amdgcn_s_wait_asynccnt)
#define HAVE_ASYNC 1
#endif
#endif
#ifndef HAVE_ASYNC
#define HAVE_ASYNC 0
#endif

#if HAVE_ASYNC
__device__ __forceinline__ void lds_async16(const _Float16* g, _Float16* l)
{
    // builtin takes generic v4i32 pointers; backend inserts the
    // global / LDS address-space casts
    __builtin_amdgcn_global_load_async_to_lds_b128(
        (v4i_*)(uintptr_t)g, (v4i_*)l, 0, 0);
}
#endif

// ---------------------------------------------------------------------------
// Core pair-MLP GEMM: one wave = one 16-row tile, NCT 16-col tiles, K staged
// through a double-buffered LDS slice (async DMA on gfx1250).
// MODE 0: A = |x_i - x_j| built on the fly from f16 node features.
// MODE 1: A = leaky(scale*h + shift) (fused BN of previous sub-layer).
// Emits f16 pre-activations + per-(q,channel) sum/sumsq stats.
// ---------------------------------------------------------------------------
template<int NCT, int MODE>
__global__ __launch_bounds__(256) void k_gemm(
    const _Float16* __restrict__ Aprev, int ldA,
    const float*    __restrict__ ss,
    const _Float16* __restrict__ nodesH,
    const _Float16* __restrict__ Wh, int Kpad,
    _Float16*       __restrict__ Out, int ldOut,
    float*          __restrict__ stats, int q0)
{
    constexpr int COUT = NCT * 16;
    __shared__ __align__(16) _Float16 sW[2][COUT * 32];
    __shared__ float sSum[COUT], sSq[COUT];
    __shared__ float sScale[MODE ? 192 : 1], sShift[MODE ? 192 : 1];

    const int tid  = threadIdx.x;
    const int wave = tid >> 5, lane = tid & 31;
    const int ql   = blockIdx.y;
    const int q    = q0 + ql;
    const int tile = blockIdx.x * 8 + wave;
    const bool act = (tile < TPQ);

    for (int i = tid; i < COUT; i += 256) { sSum[i] = 0.f; sSq[i] = 0.f; }
    if (MODE == 1) {
        for (int i = tid; i < Kpad; i += 256) {
            sScale[i] = ss[q * 192 + i];
            sShift[i] = ss[SSOFF + q * 192 + i];
        }
    }

    v8f acc[NCT];
    v8f zero = {0.f,0.f,0.f,0.f,0.f,0.f,0.f,0.f};
#pragma unroll
    for (int ct = 0; ct < NCT; ++ct) acc[ct] = zero;

    const int m    = lane & 15;
    const int off  = (lane < 16) ? 0 : 8;
    const int rloc = tile * 16 + m;

    const _Float16 *xi = nullptr, *xj = nullptr, *ap = nullptr;
    if (act) {
        if (MODE == 0) {
            int b   = rloc / NN;
            int rem = rloc - b * NN;
            int i   = rem / N_;
            int j   = rem - i * N_;
            xi = nodesH + ((size_t)(b * Q_ + q) * N_ + i) * FSTR;
            xj = nodesH + ((size_t)(b * Q_ + q) * N_ + j) * FSTR;
        } else {
            ap = Aprev + (size_t)(ql * RPQ + rloc) * ldA;
        }
    }

    // stage one 32-wide K slice of the weights into LDS buffer `buf`
    auto stage = [&](int kt, int buf) {
        for (int idx = tid; idx < COUT * 4; idx += 256) {
            int c = idx >> 2, part = idx & 3;
#if HAVE_ASYNC
            lds_async16(Wh + (size_t)c * Kpad + kt * 32 + part * 8,
                        &sW[buf][idx * 8]);
#else
            reinterpret_cast<uint4*>(&sW[buf][0])[idx] =
                reinterpret_cast<const uint4*>(Wh + (size_t)c * Kpad + kt * 32)[part];
#endif
        }
    };

    const int nk = Kpad >> 5;
    stage(0, 0);
    for (int kt = 0; kt < nk; ++kt) {
#if HAVE_ASYNC
        __builtin_amdgcn_s_wait_asynccnt(0);
#endif
        __syncthreads();                 // slice kt visible; prev reads done
        if (kt + 1 < nk) stage(kt + 1, (kt + 1) & 1);   // overlap with compute
        const _Float16* wS = &sW[kt & 1][0];
        if (act) {
            v16h a;
#pragma unroll
            for (int h = 0; h < 2; ++h) {
                const int kb = kt * 32 + off + h * 16;
                if (MODE == 0) {
                    uint4 ux = *reinterpret_cast<const uint4*>(xi + kb);
                    uint4 uy = *reinterpret_cast<const uint4*>(xj + kb);
                    const _Float16* px = reinterpret_cast<const _Float16*>(&ux);
                    const _Float16* py = reinterpret_cast<const _Float16*>(&uy);
#pragma unroll
                    for (int e = 0; e < 8; ++e)
                        a[h * 8 + e] = (_Float16)fabsf((float)px[e] - (float)py[e]);
                } else {
                    uint4 uh = *reinterpret_cast<const uint4*>(ap + kb);
                    const _Float16* ph = reinterpret_cast<const _Float16*>(&uh);
#pragma unroll
                    for (int e = 0; e < 8; ++e) {
                        int ch = kb + e;
                        float v = sScale[ch] * (float)ph[e] + sShift[ch];
                        v = (v > 0.f) ? v : LREL * v;
                        a[h * 8 + e] = (_Float16)v;
                    }
                }
            }
            const int nsel = (lane < 16) ? 0 : 8;
            const int n    = lane & 15;
#pragma unroll
            for (int ct = 0; ct < NCT; ++ct) {
                const _Float16* wp = wS + ((ct * 16 + n) * 32 + nsel);
                uint4 u0 = reinterpret_cast<const uint4*>(wp)[0];
                uint4 u1 = reinterpret_cast<const uint4*>(wp)[2];
                const _Float16* p0 = reinterpret_cast<const _Float16*>(&u0);
                const _Float16* p1 = reinterpret_cast<const _Float16*>(&u1);
                v16h bf;
#pragma unroll
                for (int e = 0; e < 8; ++e) { bf[e] = p0[e]; bf[8 + e] = p1[e]; }
                acc[ct] = __builtin_amdgcn_wmma_f32_16x16x32_f16(
                    false, a, false, bf, (short)0, acc[ct], false, false);
            }
        }
    }

    if (act) {
        const int n  = lane & 15;
        const int mb = (lane < 16) ? 0 : 8;
#pragma unroll
        for (int ct = 0; ct < NCT; ++ct) {
            float s = 0.f, s2 = 0.f;
#pragma unroll
            for (int v = 0; v < 8; ++v) {
                float d = acc[ct][v];
                s += d; s2 += d * d;
                size_t row = (size_t)(ql * RPQ + tile * 16 + mb + v);
                Out[row * ldOut + ct * 16 + n] = (_Float16)d;
            }
            s  += __shfl_xor(s,  16, 32);
            s2 += __shfl_xor(s2, 16, 32);
            if (lane < 16) {
                atomicAdd(&sSum[ct * 16 + n], s);
                atomicAdd(&sSq [ct * 16 + n], s2);
            }
        }
    }
    __syncthreads();
    for (int i = tid; i < COUT; i += 256) {
        atomicAdd(&stats[q * 192 + i], sSum[i]);
        atomicAdd(&stats[SSOFF + q * 192 + i], sSq[i]);
    }
}

// BN stats -> fused (scale, shift) per (q, channel)
__global__ void k_finalize(const float* __restrict__ stats,
                           const float* __restrict__ gamma,
                           const float* __restrict__ beta,
                           float* __restrict__ ss,
                           int Cout, int q0, int nq, float inv_cnt)
{
    int t = blockIdx.x * blockDim.x + threadIdx.x;
    if (t >= nq * Cout) return;
    int ql = t / Cout, c = t - ql * Cout;
    int q  = q0 + ql;
    float mean = stats[q * 192 + c] * inv_cnt;
    float var  = stats[SSOFF + q * 192 + c] * inv_cnt - mean * mean;
    float rs   = rsqrtf(var + EPS_);
    float sc   = gamma[c] * rs;
    ss[q * 192 + c]         = sc;
    ss[SSOFF + q * 192 + c] = beta[c] - mean * sc;
}

// final 96 -> 1 of the pair MLP (bias dropped: softmax shift-invariant)
__global__ void k_logits(const _Float16* __restrict__ H,
                         const float* __restrict__ ss3,
                         const float* __restrict__ Wl,
                         float* __restrict__ logits, int q0)
{
    int r = blockIdx.x * blockDim.x + threadIdx.x;
    if (r >= QC * RPQ) return;
    int q = q0 + r / RPQ;
    const _Float16* h = H + (size_t)r * 96;
    float sum = 0.f;
    for (int c = 0; c < 96; ++c) {
        float v = ss3[q * 192 + c] * (float)h[c] + ss3[SSOFF + q * 192 + c];
        v = (v > 0.f) ? v : LREL * v;
        sum += v * Wl[c];
    }
    logits[(size_t)q0 * RPQ + r] = sum;
}

// diagonal-masked softmax over the 26 neighbours (in place in logits buf)
__global__ void k_softmax(float* __restrict__ logits, int q0)
{
    int t = blockIdx.x * blockDim.x + threadIdx.x;
    if (t >= QC * B_ * N_) return;
    int i  = t % N_;
    int b  = (t / N_) % B_;
    int ql = t / (N_ * B_);
    size_t base = (size_t)(q0 + ql) * RPQ + (size_t)b * NN + (size_t)i * N_;
    float v[N_];
    float mx = -1e30f;
    for (int j = 0; j < N_; ++j) {
        float x = logits[base + j] - ((j == i) ? 1e8f : 0.f);
        v[j] = x; mx = fmaxf(mx, x);
    }
    float s = 0.f;
    for (int j = 0; j < N_; ++j) { v[j] = __expf(v[j] - mx); s += v[j]; }
    float inv = 1.f / s;
    for (int j = 0; j < N_; ++j) logits[base + j] = v[j] * inv;
}

// gconv (layers 0/1): y = concat(x_i, m@x) @ W^T, pre-BN (bias dropped)
__global__ void k_gconv(const float* __restrict__ nodes,
                        const float* __restrict__ m,
                        const float* __restrict__ W, int fin,
                        float* __restrict__ Hgc,
                        float* __restrict__ gstats)
{
    __shared__ float agg[FSTR];
    int bx  = blockIdx.x;
    int q   = bx / (B_ * N_);
    int rem = bx - q * (B_ * N_);
    int b   = rem / N_;
    int i   = rem - b * N_;
    const float* mb = m + (size_t)q * RPQ + (size_t)b * NN + (size_t)i * N_;
    const float* xb = nodes + ((size_t)(b * Q_ + q) * N_) * FSTR;
    for (int f = threadIdx.x; f < fin; f += blockDim.x) {
        float s = 0.f;
        for (int j = 0; j < N_; ++j) s += mb[j] * xb[(size_t)j * FSTR + f];
        agg[f] = s;
    }
    __syncthreads();
    const float* xi = xb + (size_t)i * FSTR;
    for (int c = threadIdx.x; c < 48; c += blockDim.x) {
        const float* w = W + (size_t)c * 2 * fin;
        float y = 0.f;
        for (int f = 0; f < fin; ++f) y += xi[f]  * w[f];
        for (int f = 0; f < fin; ++f) y += agg[f] * w[fin + f];
        Hgc[((size_t)(q * B_ + b) * N_ + i) * 48 + c] = y;
        atomicAdd(&gstats[q * 192 + c], y);
        atomicAdd(&gstats[SSOFF + q * 192 + c], y * y);
    }
}

// apply gconv BN + leaky and append new features to both node buffers
__global__ void k_concat(const float* __restrict__ Hgc,
                         const float* __restrict__ gss,
                         float* __restrict__ nodesF,
                         _Float16* __restrict__ nodesH, int fin)
{
    int t = blockIdx.x * blockDim.x + threadIdx.x;
    if (t >= Q_ * B_ * N_ * 48) return;
    int c = t % 48;
    int n = (t / 48) % N_;
    int b = (t / (48 * N_)) % B_;
    int q = t / (48 * N_ * B_);
    float y = Hgc[((size_t)(q * B_ + b) * N_ + n) * 48 + c];
    float v = gss[q * 192 + c] * y + gss[SSOFF + q * 192 + c];
    v = (v > 0.f) ? v : LREL * v;
    size_t o = ((size_t)(b * Q_ + q) * N_ + n) * FSTR + fin + c;
    nodesF[o] = v;
    nodesH[o] = (_Float16)v;
}

// last gconv: only the query node (i=0), bias kept, writes [B,Q,5]
__global__ void k_gconv_last(const float* __restrict__ nodes,
                             const float* __restrict__ m,
                             const float* __restrict__ W,
                             const float* __restrict__ bias,
                             float* __restrict__ out)
{
    __shared__ float agg[232];
    int q = blockIdx.x / B_;
    int b = blockIdx.x - q * B_;
    const int fin = 229;
    const float* mb = m + (size_t)q * RPQ + (size_t)b * NN;   // row i = 0
    const float* xb = nodes + ((size_t)(b * Q_ + q) * N_) * FSTR;
    for (int f = threadIdx.x; f < fin; f += blockDim.x) {
        float s = 0.f;
        for (int j = 0; j < N_; ++j) s += mb[j] * xb[(size_t)j * FSTR + f];
        agg[f] = s;
    }
    __syncthreads();
    const float* xi = xb;                                      // node 0
    for (int c = threadIdx.x; c < NW; c += blockDim.x) {
        const float* w = W + (size_t)c * 2 * fin;
        float y = bias[c];
        for (int f = 0; f < fin; ++f) y += xi[f]  * w[f];
        for (int f = 0; f < fin; ++f) y += agg[f] * w[fin + f];
        out[((size_t)b * Q_ + q) * NW + c] = y;
    }
}

// build initial node features (f32 + f16 copies, zero padded to FSTR)
__global__ void k_build_nodes(const float* __restrict__ se,
                              const float* __restrict__ oh,
                              const float* __restrict__ qe,
                              float* __restrict__ nodesF,
                              _Float16* __restrict__ nodesH)
{
    int t = blockIdx.x * blockDim.x + threadIdx.x;
    if (t >= B_ * Q_ * N_ * FSTR) return;
    int k = t & (FSTR - 1);
    int n = (t >> 8) % N_;
    int q = (t / (FSTR * N_)) % Q_;
    int b = t / (FSTR * N_ * Q_);
    float v = 0.f;
    if (n == 0) {
        if (k < 128)      v = qe[((size_t)b * Q_ + q) * 128 + k];
        else if (k < 133) v = 0.2f;                     // 1/NUM_WAYS pseudo-label
    } else {
        int s = n - 1;
        if (k < 128)      v = se[((size_t)b * S_ + s) * 128 + k];
        else if (k < 133) v = oh[((size_t)b * S_ + s) * NW + (k - 128)];
    }
    nodesF[t] = v;
    nodesH[t] = (_Float16)v;
}

// f32 [Cout x K] weights -> zero-padded f16 [Cout x Kpad]
__global__ void k_convW(const float* __restrict__ src, _Float16* __restrict__ dst,
                        int Cout, int K, int Kpad)
{
    int t = blockIdx.x * blockDim.x + threadIdx.x;
    if (t >= Cout * Kpad) return;
    int c = t / Kpad, k = t - c * Kpad;
    dst[t] = (k < K) ? (_Float16)src[(size_t)c * K + k] : (_Float16)0.f;
}

// ---------------------------------------------------------------------------
extern "C" void kernel_launch(void* const* d_in, const int* in_sizes, int n_in,
                              void* d_out, int out_size, void* d_ws, size_t ws_size,
                              hipStream_t stream)
{
    (void)in_sizes; (void)n_in; (void)out_size; (void)ws_size;
    const float* se = (const float*)d_in[0];
    const float* oh = (const float*)d_in[1];
    const float* qe = (const float*)d_in[2];
    float* out = (float*)d_out;

    char* ws = (char*)d_ws;
    size_t off = 0;
    auto alloc = [&](size_t bytes) -> char* {
        char* p = ws + off; off += (bytes + 255) & ~(size_t)255; return p;
    };

    float*    nodesF = (float*)   alloc((size_t)B_ * Q_ * N_ * FSTR * 4);
    _Float16* nodesH = (_Float16*)alloc((size_t)B_ * Q_ * N_ * FSTR * 2);
    _Float16* Ha     = (_Float16*)alloc((size_t)QC * RPQ * 192 * 2);
    _Float16* Hb     = (_Float16*)alloc((size_t)QC * RPQ * 192 * 2);
    float*    lg     = (float*)   alloc((size_t)Q_ * RPQ * 4);
    float*    Hgc    = (float*)   alloc((size_t)Q_ * B_ * N_ * 48 * 4);
    float*    stats  = (float*)   alloc((size_t)5 * Q_ * 192 * 2 * 4);
    float*    ssb    = (float*)   alloc((size_t)5 * Q_ * 192 * 2 * 4);
    _Float16* Wh[3];
    for (int L = 0; L < 3; ++L) Wh[L] = (_Float16*)alloc((size_t)113664 * 2);

    // ---- parameter leaf indices (JAX pytree: sorted dict keys) ----
    const int Kpad1[3]    = {160, 192, 256};
    const int finL[3]     = {133, 181, 229};
    const int WsIdx[3][4] = {{28,29,30,31},{50,51,52,53},{6,7,8,9}};
    const int Kin[3][4]   = {{133,192,192,96},{181,192,192,96},{229,192,192,96}};
    const int CoutA[4]    = {192,192,96,96};
    const int WlIdx[3]    = {27,49,5};
    const int GamBase[3]  = {41,63,19};
    const int BetBase[3]  = {33,55,11};
    const int gcWI[3] = {23,45,3}, gcB[3]={24,46,4}, gcBeta[2] = {25,47}, gcGam[2] = {26,48};
    const size_t Woff[4]  = {0, 49152, 86016, 104448};

    for (int L = 0; L < 3; ++L)
        for (int s = 0; s < 4; ++s) {
            int Kp  = (s == 0) ? Kpad1[L] : ((s < 3) ? 192 : 96);
            int tot = CoutA[s] * Kp;
            k_convW<<<(tot + 255) / 256, 256, 0, stream>>>(
                (const float*)d_in[WsIdx[L][s]], Wh[L] + Woff[s], CoutA[s], Kin[L][s], Kp);
        }

    {
        int tot = B_ * Q_ * N_ * FSTR;
        k_build_nodes<<<(tot + 255) / 256, 256, 0, stream>>>(se, oh, qe, nodesF, nodesH);
    }

    const float inv_pair = 1.f / (float)RPQ;
    const float inv_gc   = 1.f / (float)(B_ * N_);

    for (int L = 0; L < 3; ++L) {
        (void)hipMemsetAsync(stats, 0, (size_t)5 * Q_ * 192 * 2 * 4, stream);
        float *st[5], *ssp[5];
        for (int k = 0; k < 5; ++k) {
            st[k]  = stats + (size_t)k * Q_ * 192 * 2;
            ssp[k] = ssb   + (size_t)k * Q_ * 192 * 2;
        }
        for (int qc = 0; qc < NCHUNK; ++qc) {
            int q0 = qc * QC;
            dim3 grid(85, QC);
            k_gemm<12,0><<<grid,256,0,stream>>>(nullptr, 0, nullptr, nodesH,
                Wh[L] + Woff[0], Kpad1[L], Ha, 192, st[0], q0);
            k_finalize<<<(QC*192+255)/256,256,0,stream>>>(st[0],
                (const float*)d_in[GamBase[L]+0], (const float*)d_in[BetBase[L]+0],
                ssp[0], 192, q0, QC, inv_pair);
            k_gemm<12,1><<<grid,256,0,stream>>>(Ha, 192, ssp[0], nullptr,
                Wh[L] + Woff[1], 192, Hb, 192, st[1], q0);
            k_finalize<<<(QC*192+255)/256,256,0,stream>>>(st[1],
                (const float*)d_in[GamBase[L]+1], (const float*)d_in[BetBase[L]+1],
                ssp[1], 192, q0, QC, inv_pair);
            k_gemm<6,1><<<grid,256,0,stream>>>(Hb, 192, ssp[1], nullptr,
                Wh[L] + Woff[2], 192, Ha, 96, st[2], q0);
            k_finalize<<<(QC*96+255)/256,256,0,stream>>>(st[2],
                (const float*)d_in[GamBase[L]+2], (const float*)d_in[BetBase[L]+2],
                ssp[2], 96, q0, QC, inv_pair);
            k_gemm<6,1><<<grid,256,0,stream>>>(Ha, 96, ssp[2], nullptr,
                Wh[L] + Woff[3], 96, Hb, 96, st[3], q0);
            k_finalize<<<(QC*96+255)/256,256,0,stream>>>(st[3],
                (const float*)d_in[GamBase[L]+3], (const float*)d_in[BetBase[L]+3],
                ssp[3], 96, q0, QC, inv_pair);
            k_logits<<<(QC*RPQ+255)/256,256,0,stream>>>(Hb, ssp[3],
                (const float*)d_in[WlIdx[L]], lg, q0);
            k_softmax<<<(QC*B_*N_+63)/64,64,0,stream>>>(lg, q0);
        }
        if (L < 2) {
            k_gconv<<<Q_*B_*N_,64,0,stream>>>(nodesF, lg,
                (const float*)d_in[gcWI[L]], finL[L], Hgc, st[4]);
            k_finalize<<<(Q_*48+255)/256,256,0,stream>>>(st[4],
                (const float*)d_in[gcGam[L]], (const float*)d_in[gcBeta[L]],
                ssp[4], 48, 0, Q_, inv_gc);
            int tot = Q_ * B_ * N_ * 48;
            k_concat<<<(tot+255)/256,256,0,stream>>>(Hgc, ssp[4], nodesF, nodesH, finL[L]);
        } else {
            k_gconv_last<<<Q_*B_,64,0,stream>>>(nodesF, lg,
                (const float*)d_in[3], (const float*)d_in[gcB[2]], out);
        }
    }
}
